// GNN_85761906966862
// MI455X (gfx1250) — matrix operand order repaired and checked by reading, not compiled, not run on hardware
//
#include <hip/hip_runtime.h>
#include <math.h>

typedef __attribute__((ext_vector_type(16))) _Float16 v16h;
typedef __attribute__((ext_vector_type(8)))  _Float16 v8h;
typedef __attribute__((ext_vector_type(8)))  float    v8f;

#define NB    16
#define NPER  1024
#define EPER  8192
#define ETOT  (NB*EPER)      /* 131072 */
#define KP1   922
#define KP2   830
#define N0    (NB*NPER)      /* 16384 */
#define N1    (NB*KP1)       /* 14752 */
#define N2    (NB*KP2)       /* 13280 */

// ---------------------------------------------------------------- elementwise
__global__ void gnn_cast_f16(const float* __restrict__ in, _Float16* __restrict__ out, int n) {
    int i = blockIdx.x * blockDim.x + threadIdx.x;
    if (i < n) out[i] = (_Float16)in[i];
}

__global__ void gnn_wt_cast(const float* __restrict__ W, _Float16* __restrict__ Wt, int K, int N) {
    int i = blockIdx.x * blockDim.x + threadIdx.x;
    if (i >= K * N) return;
    int k = i / N, n = i % N;
    Wt[(size_t)n * K + k] = (_Float16)W[i];
}

__global__ void gnn_fill_f32(float* p, float v, int n) {
    int i = blockIdx.x * blockDim.x + threadIdx.x;
    if (i < n) p[i] = v;
}

__global__ void gnn_fill_i32(int* p, int v, int n) {
    int i = blockIdx.x * blockDim.x + threadIdx.x;
    if (i < n) p[i] = v;
}

__global__ void gnn_relu(float* p, int n) {
    int i = blockIdx.x * blockDim.x + threadIdx.x;
    if (i < n) p[i] = fmaxf(p[i], 0.0f);
}

__global__ void gnn_rsqrt(float* p, int n) {
    int i = blockIdx.x * blockDim.x + threadIdx.x;
    if (i < n) p[i] = rsqrtf(p[i]);
}

// ---------------------------------------------------------------- WMMA GEMM
// C[M,N] = A[M,K](f16,row-major) * Bt[N,K](f16, B transposed), f32 accumulate.
// One wave computes a 2x4 block of 16x16 tiles (32 rows x 64 cols). All 12
// b128 loads of a K-step are issued before the 8 WMMAs so the scheduler can
// clause them and overlap the next iteration's loads with the matrix ops.
// Requires M%32==0, N%64==0, K%32==0 (true for every layer here).
__global__ void gnn_gemm_wmma(const _Float16* __restrict__ A,
                              const _Float16* __restrict__ Bt,
                              float* __restrict__ C, int M, int N, int K) {
    int wavesPerBlk = blockDim.x >> 5;
    int w    = blockIdx.x * wavesPerBlk + (threadIdx.x >> 5);
    int lane = threadIdx.x & 31;
    int Mt2 = M >> 5;                 // groups of 2 M-tiles
    int Nt4 = N >> 6;                 // groups of 4 N-tiles
    if (w >= Mt2 * Nt4) return;       // wave-uniform exit (EXEC stays full inside)
    int tm2 = w / Nt4, tn4 = w % Nt4;

    int hi   = lane >> 4;             // 0 for lanes 0-15, 1 for 16-31
    int aoff = hi ? 16 : 0;           // A: per-lane-half K subrange
    int boff = hi ? 16 : 0;           // B: per-lane-half K subrange
    size_t bstride = (size_t)16 * K;  // between adjacent N-tiles in Bt

    const _Float16* Ap0 = A + (size_t)(tm2 * 32 + (lane & 15)) * K + aoff;
    const _Float16* Ap1 = Ap0 + (size_t)16 * K;
    const _Float16* Bp0 = Bt + (size_t)(tn4 * 64 + (lane & 15)) * K + boff;
    const _Float16* Bp1 = Bp0 + bstride;
    const _Float16* Bp2 = Bp1 + bstride;
    const _Float16* Bp3 = Bp2 + bstride;

    v8f acc[2][4] = {};
    for (int k0 = 0; k0 < K; k0 += 32) {
        union { v16h v; v8h h[2]; } a0, a1;
        a0.h[0] = *(const v8h*)(Ap0 + k0);        // K = k0+aoff .. +7
        a0.h[1] = *(const v8h*)(Ap0 + k0 + 8);    // K = k0+aoff+8 .. +15
        a1.h[0] = *(const v8h*)(Ap1 + k0);
        a1.h[1] = *(const v8h*)(Ap1 + k0 + 8);
        v16h b0 = *(const v16h*)(Bp0 + k0);       // 16 contiguous K values
        v16h b1 = *(const v16h*)(Bp1 + k0);
        v16h b2 = *(const v16h*)(Bp2 + k0);
        v16h b3 = *(const v16h*)(Bp3 + k0);
        acc[0][0] = __builtin_amdgcn_wmma_f32_16x16x32_f16(false, a0.v, false, b0, (short)0, acc[0][0], false, false);
        acc[1][0] = __builtin_amdgcn_wmma_f32_16x16x32_f16(false, a1.v, false, b0, (short)0, acc[1][0], false, false);
        acc[0][1] = __builtin_amdgcn_wmma_f32_16x16x32_f16(false, a0.v, false, b1, (short)0, acc[0][1], false, false);
        acc[1][1] = __builtin_amdgcn_wmma_f32_16x16x32_f16(false, a1.v, false, b1, (short)0, acc[1][1], false, false);
        acc[0][2] = __builtin_amdgcn_wmma_f32_16x16x32_f16(false, a0.v, false, b2, (short)0, acc[0][2], false, false);
        acc[1][2] = __builtin_amdgcn_wmma_f32_16x16x32_f16(false, a1.v, false, b2, (short)0, acc[1][2], false, false);
        acc[0][3] = __builtin_amdgcn_wmma_f32_16x16x32_f16(false, a0.v, false, b3, (short)0, acc[0][3], false, false);
        acc[1][3] = __builtin_amdgcn_wmma_f32_16x16x32_f16(false, a1.v, false, b3, (short)0, acc[1][3], false, false);
    }
    int ccol0 = tn4 * 64 + (lane & 15);
#pragma unroll
    for (int i = 0; i < 2; ++i) {
        int crow0 = tm2 * 32 + i * 16 + (hi << 3);
#pragma unroll
        for (int j = 0; j < 4; ++j) {
#pragma unroll
            for (int r = 0; r < 8; ++r)
                C[(size_t)(crow0 + r) * N + ccol0 + j * 16] = acc[i][j][r];
        }
    }
}

// ---------------------------------------------------------------- GCN pieces
__global__ void gnn_deg(const int* __restrict__ dst, const int* __restrict__ m,
                        float* deg, int E) {
    int e = blockIdx.x * blockDim.x + threadIdx.x;
    if (e >= E) return;
    if (m[e])
        (void)__hip_atomic_fetch_add(&deg[dst[e]], 1.0f,
                                     __ATOMIC_RELAXED, __HIP_MEMORY_SCOPE_AGENT);
}

// out[n,c] = dinv[n]^2 * xw[n,c] + b[c]   (self-loop + bias baseline for scatter)
__global__ void gnn_selfloop(const float* __restrict__ xw, const float* __restrict__ dinv,
                             const float* __restrict__ bias, float* __restrict__ out,
                             int n, int C) {
    int i = blockIdx.x * blockDim.x + threadIdx.x;
    if (i >= n * C) return;
    int node = i / C, c = i % C;
    float di = dinv[node];
    out[i] = di * di * xw[i] + bias[c];
}

// one wave per edge; lanes stride channels with float4; atomic f32 adds into agg[dst]
__global__ void gnn_scatter(const int* __restrict__ src, const int* __restrict__ dst,
                            const int* __restrict__ m, const float* __restrict__ dinv,
                            const float* __restrict__ xw, float* __restrict__ agg,
                            int E, int C) {
    int e    = (blockIdx.x * blockDim.x + threadIdx.x) >> 5;
    int lane = threadIdx.x & 31;
    if (e >= E) return;
    if (!m[e]) return;                               // wave-uniform
    int s = src[e], d = dst[e];
    float nrm = dinv[s] * dinv[d];
    const float4* xs = (const float4*)(xw + (size_t)s * C);
    float* ad = agg + (size_t)d * C;
    int C4 = C >> 2;
    for (int c4 = lane; c4 < C4; c4 += 32) {
        float4 v = xs[c4];
        int c = c4 << 2;
        (void)__hip_atomic_fetch_add(&ad[c + 0], nrm * v.x, __ATOMIC_RELAXED, __HIP_MEMORY_SCOPE_AGENT);
        (void)__hip_atomic_fetch_add(&ad[c + 1], nrm * v.y, __ATOMIC_RELAXED, __HIP_MEMORY_SCOPE_AGENT);
        (void)__hip_atomic_fetch_add(&ad[c + 2], nrm * v.z, __ATOMIC_RELAXED, __HIP_MEMORY_SCOPE_AGENT);
        (void)__hip_atomic_fetch_add(&ad[c + 3], nrm * v.w, __ATOMIC_RELAXED, __HIP_MEMORY_SCOPE_AGENT);
    }
}

// ---------------------------------------------------------------- pooling
__global__ void gnn_pnorm(const float* __restrict__ p, int n, float* out) {
    __shared__ float red[256];
    float s = 0.0f;
    for (int i = threadIdx.x; i < n; i += 256) s += p[i] * p[i];
    red[threadIdx.x] = s;
    __syncthreads();
    for (int off = 128; off > 0; off >>= 1) {
        if ((int)threadIdx.x < off) red[threadIdx.x] += red[threadIdx.x + off];
        __syncthreads();
    }
    if (threadIdx.x == 0) out[0] = sqrtf(red[0]);
}

// one wave per node: sc[n] = h[n,:] . p
__global__ void gnn_score(const float* __restrict__ h, const float* __restrict__ p,
                          float* __restrict__ sc, int n, int C) {
    int w    = (blockIdx.x * blockDim.x + threadIdx.x) >> 5;
    int lane = threadIdx.x & 31;
    if (w >= n) return;
    const float* row = h + (size_t)w * C;
    float s = 0.0f;
    for (int c = lane; c < C; c += 32) s += row[c] * p[c];
    for (int o = 16; o > 0; o >>= 1) s += __shfl_down(s, o, 32);
    if (lane == 0) sc[w] = s;
}

// per-graph bitonic sort (descending) over 1024 LDS slots, keep top-k
__global__ void gnn_topk(const float* __restrict__ sc, const float* __restrict__ pnorm,
                         int n_per, int k, int* __restrict__ new_id,
                         int* __restrict__ perm, float* __restrict__ gate) {
    int g = blockIdx.x;
    __shared__ float sk[1024];
    __shared__ int   si[1024];
    for (int t = threadIdx.x; t < 1024; t += blockDim.x) {
        if (t < n_per) { sk[t] = sc[g * n_per + t]; si[t] = t; }
        else           { sk[t] = -3.4e38f;          si[t] = t; }
    }
    __syncthreads();
    for (int kk = 2; kk <= 1024; kk <<= 1) {
        for (int j = kk >> 1; j > 0; j >>= 1) {
            for (int t = threadIdx.x; t < 512; t += blockDim.x) {
                int i = ((t & ~(j - 1)) << 1) | (t & (j - 1));
                int p = i | j;
                bool up = ((i & kk) == 0);
                float a = sk[i], b = sk[p];
                if ((a < b) == up) {
                    sk[i] = b; sk[p] = a;
                    int ti = si[i]; si[i] = si[p]; si[p] = ti;
                }
            }
            __syncthreads();
        }
    }
    float pn = pnorm[0];
    for (int t = threadIdx.x; t < k; t += blockDim.x) {
        int node = g * n_per + si[t];
        int nid  = g * k + t;
        new_id[node] = nid;
        perm[nid]    = node;
        gate[nid]    = tanhf(sk[t] / pn);
    }
}

__global__ void gnn_gather(const float* __restrict__ h, const int* __restrict__ perm,
                           const float* __restrict__ gate, float* __restrict__ out,
                           int nNew, int C) {
    int i = blockIdx.x * blockDim.x + threadIdx.x;
    if (i >= nNew * C) return;
    int node = i / C, c = i % C;
    out[i] = h[(size_t)perm[node] * C + c] * gate[node];
}

__global__ void gnn_remap(const int* __restrict__ s0, const int* __restrict__ d0,
                          const int* __restrict__ m0, const int* __restrict__ new_id,
                          int* __restrict__ s1, int* __restrict__ d1, int* __restrict__ m1,
                          int E) {
    int e = blockIdx.x * blockDim.x + threadIdx.x;
    if (e >= E) return;
    int ns = new_id[s0[e]], nd = new_id[d0[e]];
    int valid = (m0[e] != 0) & (ns >= 0) & (nd >= 0);
    s1[e] = valid ? ns : 0;
    d1[e] = valid ? nd : 0;
    m1[e] = valid;
}

// ---------------------------------------------------------------- readout
__global__ void gnn_mean(const float* __restrict__ h, float* __restrict__ g,
                         int Bn, int K, int C) {
    int i = blockIdx.x * blockDim.x + threadIdx.x;
    if (i >= Bn * C) return;
    int b = i / C, c = i % C;
    const float* base = h + ((size_t)b * K) * C + c;
    float s = 0.0f;
    for (int t = 0; t < K; ++t) s += base[(size_t)t * C];
    g[i] = s / (float)K;
}

// logits = g @ Wlin + blin ; log_softmax row-wise. 16x16 output, one block.
__global__ void gnn_final(const float* __restrict__ g, const float* __restrict__ Wl,
                          const float* __restrict__ bl, float* __restrict__ out) {
    __shared__ float lg[16][16];
    __shared__ float rmax[16], rlse[16];
    int t = threadIdx.x;          // 256 threads
    int b = t >> 4, oc = t & 15;
    const float* gr = g + (size_t)b * 1024;
    float s = bl[oc];
    for (int c = 0; c < 1024; ++c) s += gr[c] * Wl[c * 16 + oc];
    lg[b][oc] = s;
    __syncthreads();
    if (oc == 0) {
        float mx = -3.4e38f;
        for (int j = 0; j < 16; ++j) mx = fmaxf(mx, lg[b][j]);
        float se = 0.0f;
        for (int j = 0; j < 16; ++j) se += expf(lg[b][j] - mx);
        rmax[b] = mx; rlse[b] = logf(se);
    }
    __syncthreads();
    out[t] = lg[b][oc] - rmax[b] - rlse[b];
}

// ---------------------------------------------------------------- host
static inline int cdiv(int a, int b) { return (a + b - 1) / b; }

extern "C" void kernel_launch(void* const* d_in, const int* in_sizes, int n_in,
                              void* d_out, int out_size, void* d_ws, size_t ws_size,
                              hipStream_t stream) {
    (void)in_sizes; (void)n_in; (void)out_size; (void)ws_size;

    const float* x    = (const float*)d_in[0];
    const int*   src0 = (const int*)  d_in[1];
    const int*   dst0 = (const int*)  d_in[2];
    const float* W1 = (const float*)d_in[3];  const float* b1 = (const float*)d_in[4];
    const float* W2 = (const float*)d_in[5];  const float* b2 = (const float*)d_in[6];
    const float* W3 = (const float*)d_in[7];  const float* b3 = (const float*)d_in[8];
    const float* p1 = (const float*)d_in[9];
    const float* W4 = (const float*)d_in[10]; const float* b4 = (const float*)d_in[11];
    const float* W5 = (const float*)d_in[12]; const float* b5 = (const float*)d_in[13];
    const float* W6 = (const float*)d_in[14]; const float* b6 = (const float*)d_in[15];
    const float* p2 = (const float*)d_in[16];
    const float* Wlin = (const float*)d_in[17]; const float* blin = (const float*)d_in[18];

    // ---- workspace carve-up
    uint8_t* ws = (uint8_t*)d_ws;
    size_t o = 0;
    auto alloc = [&](size_t bytes) -> void* {
        o = (o + 255) & ~(size_t)255;
        void* p = ws + o;
        o += bytes;
        return p;
    };
    float*     bufA  = (float*)    alloc((size_t)N0 * 1024 * 4);
    float*     bufB  = (float*)    alloc((size_t)N0 * 1024 * 4);
    _Float16*  hf16  = (_Float16*) alloc((size_t)N0 * 512 * 2);
    _Float16*  wt1   = (_Float16*) alloc((size_t)128 * 64 * 2);
    _Float16*  wt2   = (_Float16*) alloc((size_t)64 * 128 * 2);
    _Float16*  wt3   = (_Float16*) alloc((size_t)128 * 256 * 2);
    _Float16*  wt4   = (_Float16*) alloc((size_t)256 * 256 * 2);
    _Float16*  wt5   = (_Float16*) alloc((size_t)256 * 512 * 2);
    _Float16*  wt6   = (_Float16*) alloc((size_t)512 * 1024 * 2);
    float*     dinv0 = (float*)    alloc((size_t)N0 * 4);
    float*     dinv1 = (float*)    alloc((size_t)N0 * 4);
    float*     scores= (float*)    alloc((size_t)N0 * 4);
    float*     pnv   = (float*)    alloc(256);
    int*       newid = (int*)      alloc((size_t)N0 * 4);
    int*       perm  = (int*)      alloc((size_t)N0 * 4);
    float*     gate  = (float*)    alloc((size_t)N0 * 4);
    int*       m0    = (int*)      alloc((size_t)ETOT * 4);
    int*       s1    = (int*)      alloc((size_t)ETOT * 4);
    int*       d1    = (int*)      alloc((size_t)ETOT * 4);
    int*       m1    = (int*)      alloc((size_t)ETOT * 4);
    float*     gbuf  = (float*)    alloc((size_t)16 * 1024 * 4);

    // ---- helpers
    auto prepw = [&](const float* W, _Float16* Wt, int K, int N) {
        gnn_wt_cast<<<cdiv(K * N, 256), 256, 0, stream>>>(W, Wt, K, N);
    };
    auto makedinv = [&](const int* ed, const int* em, float* dinv, int n) {
        gnn_fill_f32<<<cdiv(n, 256), 256, 0, stream>>>(dinv, 1.0f, n);
        gnn_deg<<<cdiv(ETOT, 256), 256, 0, stream>>>(ed, em, dinv, ETOT);
        gnn_rsqrt<<<cdiv(n, 256), 256, 0, stream>>>(dinv, n);
    };
    auto conv = [&](const float* hin, float* hout, float* xwbuf,
                    const _Float16* Wt, const float* bias, const float* dinv,
                    const int* es, const int* ed, const int* em,
                    int n, int Cin, int Cout) {
        int nin = n * Cin, nout = n * Cout;
        gnn_cast_f16<<<cdiv(nin, 256), 256, 0, stream>>>(hin, hf16, nin);
        int tiles = (n / 32) * (Cout / 64);     // 2x4 tile blocks per wave
        gnn_gemm_wmma<<<cdiv(tiles, 8), 256, 0, stream>>>(hf16, Wt, xwbuf, n, Cout, Cin);
        gnn_selfloop<<<cdiv(nout, 256), 256, 0, stream>>>(xwbuf, dinv, bias, hout, n, Cout);
        gnn_scatter<<<cdiv(ETOT * 32, 256), 256, 0, stream>>>(es, ed, em, dinv, xwbuf, hout, ETOT, Cout);
        gnn_relu<<<cdiv(nout, 256), 256, 0, stream>>>(hout, nout);
    };
    auto pool = [&](const float* h, float* xnew, const float* p, int plen,
                    int n_per, int k, int C) {
        int n = NB * n_per;
        gnn_pnorm<<<1, 256, 0, stream>>>(p, plen, pnv);
        gnn_score<<<cdiv(n * 32, 256), 256, 0, stream>>>(h, p, scores, n, C);
        gnn_fill_i32<<<cdiv(n, 256), 256, 0, stream>>>(newid, -1, n);
        gnn_topk<<<NB, 512, 0, stream>>>(scores, pnv, n_per, k, newid, perm, gate);
        int nnew = NB * k;
        gnn_gather<<<cdiv(nnew * C, 256), 256, 0, stream>>>(h, perm, gate, xnew, nnew, C);
    };

    // ---- weight prep + edge mask
    prepw(W1, wt1, 128, 64);
    prepw(W2, wt2, 64, 128);
    prepw(W3, wt3, 128, 256);
    prepw(W4, wt4, 256, 256);
    prepw(W5, wt5, 256, 512);
    prepw(W6, wt6, 512, 1024);
    gnn_fill_i32<<<cdiv(ETOT, 256), 256, 0, stream>>>(m0, 1, ETOT);

    // ---- stage 1: conv1..3 on original graph
    makedinv(dst0, m0, dinv0, N0);
    conv(x,    bufA, bufB, wt1, b1, dinv0, src0, dst0, m0, N0, 128, 64);
    conv(bufA, bufA, bufB, wt2, b2, dinv0, src0, dst0, m0, N0, 64, 128);
    conv(bufA, bufA, bufB, wt3, b3, dinv0, src0, dst0, m0, N0, 128, 256);

    // ---- pool1: keep 922/1024 per graph; remap edges
    pool(bufA, bufB, p1, 256, NPER, KP1, 256);
    gnn_remap<<<cdiv(ETOT, 256), 256, 0, stream>>>(src0, dst0, m0, newid, s1, d1, m1, ETOT);

    // ---- stage 2: conv4..6 on pooled graph (h lives in bufB)
    makedinv(d1, m1, dinv1, N1);
    conv(bufB, bufB, bufA, wt4, b4, dinv1, s1, d1, m1, N1, 256, 256);
    conv(bufB, bufB, bufA, wt5, b5, dinv1, s1, d1, m1, N1, 256, 512);
    conv(bufB, bufB, bufA, wt6, b6, dinv1, s1, d1, m1, N1, 512, 1024);

    // ---- pool2: keep 830/922 per graph (no edge remap needed afterwards)
    pool(bufB, bufA, p2, 1024, KP1, KP2, 1024);

    // ---- readout
    gnn_mean<<<cdiv(16 * 1024, 256), 256, 0, stream>>>(bufA, gbuf, 16, KP2, 1024);
    gnn_final<<<1, 256, 0, stream>>>(gbuf, Wlin, blin, (float*)d_out);
}
